// TriMul_59665685676252
// MI455X (gfx1250) — matrix operand order repaired and checked by reading, not compiled, not run on hardware
//
#include <hip/hip_runtime.h>
#include <hip/hip_bf16.h>

typedef unsigned int   u32;
typedef unsigned short u16;
typedef __bf16 bf16;
typedef bf16  bf16x16 __attribute__((ext_vector_type(16)));
typedef float f32x8   __attribute__((ext_vector_type(8)));

#define SDIM 512
#define SS   (512 * 512)
#define DIM  128
#define LNEPS 1e-5f

// ---------- small helpers ----------
__device__ __forceinline__ u16 f2bf(float f) {
    u32 u = __float_as_uint(f);
    u32 r = u + 0x7FFFu + ((u >> 16) & 1u);     // round-to-nearest-even
    return (u16)(r >> 16);
}
__device__ __forceinline__ u32 pack2bf(float lo, float hi) {
    return (u32)f2bf(lo) | ((u32)f2bf(hi) << 16);
}
__device__ __forceinline__ float bf2f(u16 h) {
    return __uint_as_float(((u32)h) << 16);
}
__device__ __forceinline__ float sigm(float x) {
    // fast sigmoid: v_exp_f32 + v_rcp_f32 (avoid IEEE div_scale/div_fmas)
    return __builtin_amdgcn_rcpf(1.0f + __expf(-x));
}

struct Frag32B { uint4 lo, hi; };
__device__ __forceinline__ bf16x16 make_frag(uint4 lo, uint4 hi) {
    Frag32B f{lo, hi};
    return __builtin_bit_cast(bf16x16, f);
}
__device__ __forceinline__ f32x8 zero8() {
    f32x8 z = {0.f,0.f,0.f,0.f,0.f,0.f,0.f,0.f};
    return z;
}

// A: 16 rows x 128 K bf16 in LDS (pitch 136 shorts).
// B: 16 cols stored transposed [n][k], pitch 136.
// 16x16 f32 tile via 4 x v_wmma_f32_16x16x32_bf16.
__device__ __forceinline__ f32x8 mm_tile16(const u16 A[][136], int rowbase,
                                           const u16 B[][136], int colbase,
                                           int lane) {
    f32x8 acc = zero8();
    const int r   = rowbase + (lane & 15);
    const int kh  = lane >> 4;
    const int col = colbase + (lane & 15);
#pragma unroll
    for (int ks = 0; ks < 4; ++ks) {
        const int k0 = ks * 32;
        uint4 a0 = *(const uint4*)&A[r][k0 + kh * 8];
        uint4 a1 = *(const uint4*)&A[r][k0 + 16 + kh * 8];
        uint4 b0 = *(const uint4*)&B[col][k0 + kh * 8];
        uint4 b1 = *(const uint4*)&B[col][k0 + 16 + kh * 8];
        bf16x16 af  = make_frag(a0, a1);
        bf16x16 bfv = make_frag(b0, b1);
        acc = __builtin_amdgcn_wmma_f32_16x16x32_bf16(
            false, af, false, bfv, (short)0, acc, false, false);
    }
    return acc;
}

// Load one 128x128 f32 weight (row-major [k][n]) transposed bf16 -> wt[n][k].
// Packs k-pairs: one b32 LDS store per two elements.
__device__ __forceinline__ void load_w(const float* __restrict__ W,
                                       u16 wt[][136], int tid) {
#pragma unroll 4
    for (int e = 0; e < 32; ++e) {
        int lin = e * 256 + tid;            // 8192 k-pairs
        int k   = (lin >> 7) * 2;
        int n   = lin & 127;
        u32 pk = pack2bf(W[(size_t)k * DIM + n], W[(size_t)(k + 1) * DIM + n]);
        *(u32*)&wt[n][k] = pk;
    }
}
// Load a (main, gate) weight pair at once.
__device__ __forceinline__ void load_w2(const float* __restrict__ W0,
                                        const float* __restrict__ W1,
                                        u16 wt[2][128][136], int tid) {
#pragma unroll 4
    for (int e = 0; e < 32; ++e) {
        int lin = e * 256 + tid;
        int k   = (lin >> 7) * 2;
        int n   = lin & 127;
        *(u32*)&wt[0][n][k] =
            pack2bf(W0[(size_t)k * DIM + n], W0[(size_t)(k + 1) * DIM + n]);
        *(u32*)&wt[1][n][k] =
            pack2bf(W1[(size_t)k * DIM + n], W1[(size_t)(k + 1) * DIM + n]);
    }
}

// =====================================================================
// Stage 1: fused LayerNorm + 5 projections + sigmoid gates + mask.
// Writes left_t/right_t channel-major [d][p] (bf16) and gate row-major [p][d].
// 256 threads = 8 waves; each block handles 128 rows (p = i*S+k).
// =====================================================================
__global__ void __launch_bounds__(256)
trimul_stage1(const float* __restrict__ x, const float* __restrict__ mask,
              const float* __restrict__ ln_g, const float* __restrict__ ln_b,
              const float* __restrict__ Wl,  const float* __restrict__ Wr,
              const float* __restrict__ Wlg, const float* __restrict__ Wrg,
              const float* __restrict__ Wog,
              u16* __restrict__ leftT, u16* __restrict__ rightT,
              u16* __restrict__ gateO) {
    __shared__ u16 xn[128][136];       // normalized rows, bf16
    __shared__ u16 wt[2][128][136];    // (main, gate) weights, transposed bf16
    __shared__ u16 tb[16][136];        // transpose staging: [d_local][p_local]
    __shared__ float msk[128];

    const int tid   = threadIdx.x;
    const int lane  = tid & 31;
    const int wv    = tid >> 5;
    const int wbase = wv * 16;
    const size_t p_base = (size_t)blockIdx.x * 128;

    // ---- LayerNorm: 2 lanes per row, 64 f32 each ----
    {
        const int row  = tid >> 1;
        const int half = tid & 1;
        const float* xr = x + (p_base + row) * DIM + half * 64;
        float4 v[16];
        float s = 0.f, sq = 0.f;
#pragma unroll
        for (int i = 0; i < 16; ++i) {
            v[i] = ((const float4*)xr)[i];
            s  += v[i].x + v[i].y + v[i].z + v[i].w;
            sq += v[i].x * v[i].x + v[i].y * v[i].y
                + v[i].z * v[i].z + v[i].w * v[i].w;
        }
        s  += __shfl_xor(s, 1, 32);
        sq += __shfl_xor(sq, 1, 32);
        const float mu  = s * (1.f / 128.f);
        const float var = sq * (1.f / 128.f) - mu * mu;
        const float rs  = rsqrtf(var + LNEPS);
#pragma unroll
        for (int i = 0; i < 16; ++i) {
            const int c = half * 64 + i * 4;
            uint2 pk;
            pk.x = pack2bf((v[i].x - mu) * rs * ln_g[c + 0] + ln_b[c + 0],
                           (v[i].y - mu) * rs * ln_g[c + 1] + ln_b[c + 1]);
            pk.y = pack2bf((v[i].z - mu) * rs * ln_g[c + 2] + ln_b[c + 2],
                           (v[i].w - mu) * rs * ln_g[c + 3] + ln_b[c + 3]);
            *(uint2*)&xn[row][c] = pk;      // one b64 store per 4 elems
        }
        if (half == 0) msk[row] = mask[p_base + row];
    }
    __syncthreads();

    const int mh = (lane >> 4) * 8;

    // ---- two gated projections: (Wl,Wlg) -> leftT, (Wr,Wrg) -> rightT ----
    const float* Wmain[2] = {Wl, Wr};
    const float* Wgate[2] = {Wlg, Wrg};
    u16* dstT[2] = {leftT, rightT};
#pragma unroll 1
    for (int pass = 0; pass < 2; ++pass) {
        load_w2(Wmain[pass], Wgate[pass], wt, tid);
        __syncthreads();
#pragma unroll 1
        for (int nt = 0; nt < 8; ++nt) {
            f32x8 aL = mm_tile16(xn, wbase, wt[0], nt * 16, lane);
            f32x8 aG = mm_tile16(xn, wbase, wt[1], nt * 16, lane);
            float g[8];
#pragma unroll
            for (int vv = 0; vv < 8; ++vv)
                g[vv] = aL[vv] * msk[wbase + mh + vv] * sigm(aG[vv]);
            uint2 q0, q1;
            q0.x = pack2bf(g[0], g[1]); q0.y = pack2bf(g[2], g[3]);
            q1.x = pack2bf(g[4], g[5]); q1.y = pack2bf(g[6], g[7]);
            *(uint2*)&tb[lane & 15][wbase + mh]     = q0;
            *(uint2*)&tb[lane & 15][wbase + mh + 4] = q1;
            __syncthreads();
            {   // cooperative coalesced store: 16 channels x 128 p
                const int dloc = tid >> 4;
                const int p8   = (tid & 15) * 8;
                uint4 val = *(const uint4*)&tb[dloc][p8];
                *(uint4*)&dstT[pass][(size_t)(nt * 16 + dloc) * SS + p_base + p8] = val;
            }
            __syncthreads();
        }
    }

    // ---- output gate: sigmoid(xn @ Wog), row-major [p][d] ----
    load_w(Wog, wt[0], tid);
    __syncthreads();
#pragma unroll 1
    for (int nt = 0; nt < 8; ++nt) {
        f32x8 g = mm_tile16(xn, wbase, wt[0], nt * 16, lane);
#pragma unroll
        for (int vv = 0; vv < 8; ++vv) {
            const size_t prow = p_base + wbase + mh + vv;
            gateO[prow * DIM + nt * 16 + (lane & 15)] = f2bf(sigm(g[vv]));
        }
    }
}

// =====================================================================
// Stage 2: 128 independent 512x512x512 NT-GEMMs (the einsum).
// Block = 128 threads (4 waves) computes a 128x64 C tile of one channel.
// Each wave: 32 rows x 64 cols = 8 WMMA per 32-k step (2 A-frags x 4 B-frags).
// Register-staged double-buffered LDS, one barrier per k step.
// =====================================================================
__global__ void __launch_bounds__(128)
trimul_stage2(const u16* __restrict__ leftT, const u16* __restrict__ rightT,
              u16* __restrict__ outb) {
    __shared__ u16 As[2][128][40];   // pitch 40 shorts: 16B rows, conflict-free
    __shared__ u16 Bs[2][64][40];

    const int bid  = blockIdx.x;
    const int d    = bid >> 5;            // 128 channels
    const int tile = bid & 31;            // 4 x 8 tiles of 128x64
    const int ti   = tile >> 3;
    const int tj   = tile & 7;

    const u16* Lp = leftT  + (size_t)d * SS + (size_t)(ti * 128) * SDIM;
    const u16* Rp = rightT + (size_t)d * SS + (size_t)(tj * 64) * SDIM;

    const int tid  = threadIdx.x;
    const int lane = tid & 31;
    const int wv   = tid >> 5;

    f32x8 acc[2][4];
#pragma unroll
    for (int h = 0; h < 2; ++h)
#pragma unroll
        for (int nt = 0; nt < 4; ++nt) acc[h][nt] = zero8();

    const int kh  = lane >> 4;
    const int ar0 = wv * 32 + (lane & 15);     // wave owns rows [wv*32, wv*32+32)

    // register staging for global->LDS double buffer
    uint4 ga[4], gb[2];
    const int la_r  = tid >> 2;                // A: 128 rows x 32 k, 4 per thread
    const int la_c8 = (tid & 3) * 8;
    const int lb_r  = tid >> 2;                // B: 64 rows, 2 per thread
    const int lb_c8 = (tid & 3) * 8;

    // prologue: fetch k-tile 0
#pragma unroll
    for (int e = 0; e < 4; ++e)
        ga[e] = *(const uint4*)&Lp[(size_t)(e * 32 + la_r) * SDIM + la_c8];
#pragma unroll
    for (int e = 0; e < 2; ++e)
        gb[e] = *(const uint4*)&Rp[(size_t)(e * 32 + lb_r) * SDIM + lb_c8];

    int cur = 0;
#pragma unroll 1
    for (int ks = 0; ks < 16; ++ks) {
        // commit staged tile to LDS buffer `cur`
#pragma unroll
        for (int e = 0; e < 4; ++e)
            *(uint4*)&As[cur][e * 32 + la_r][la_c8] = ga[e];
#pragma unroll
        for (int e = 0; e < 2; ++e)
            *(uint4*)&Bs[cur][e * 32 + lb_r][lb_c8] = gb[e];
        __syncthreads();

        // issue next tile's global loads (overlap with WMMA below)
        if (ks + 1 < 16) {
            const int k0n = (ks + 1) * 32;
#pragma unroll
            for (int e = 0; e < 4; ++e)
                ga[e] = *(const uint4*)&Lp[(size_t)(e * 32 + la_r) * SDIM + k0n + la_c8];
#pragma unroll
            for (int e = 0; e < 2; ++e)
                gb[e] = *(const uint4*)&Rp[(size_t)(e * 32 + lb_r) * SDIM + k0n + lb_c8];
        }

        // compute: 2 A-frags x 4 B-frags
        bf16x16 af[2];
#pragma unroll
        for (int h = 0; h < 2; ++h) {
            const int r = ar0 + h * 16;
            uint4 a0 = *(const uint4*)&As[cur][r][kh * 8];
            uint4 a1 = *(const uint4*)&As[cur][r][16 + kh * 8];
            af[h] = make_frag(a0, a1);
        }
#pragma unroll
        for (int nt = 0; nt < 4; ++nt) {
            const int br = nt * 16 + (lane & 15);
            uint4 b0 = *(const uint4*)&Bs[cur][br][kh * 8];
            uint4 b1 = *(const uint4*)&Bs[cur][br][16 + kh * 8];
            bf16x16 bfv = make_frag(b0, b1);
#pragma unroll
            for (int h = 0; h < 2; ++h)
                acc[h][nt] = __builtin_amdgcn_wmma_f32_16x16x32_bf16(
                    false, af[h], false, bfv, (short)0, acc[h][nt], false, false);
        }
        cur ^= 1;
    }

    // epilogue: bf16 store to [i][j][d]
    const int mh = (lane >> 4) * 8;
#pragma unroll
    for (int h = 0; h < 2; ++h) {
#pragma unroll
        for (int nt = 0; nt < 4; ++nt) {
#pragma unroll
            for (int vv = 0; vv < 8; ++vv) {
                const int ii = ti * 128 + wv * 32 + h * 16 + mh + vv;
                const int jj = tj * 64 + nt * 16 + (lane & 15);
                outb[((size_t)ii * SDIM + jj) * DIM + d] = f2bf(acc[h][nt][vv]);
            }
        }
    }
}

// =====================================================================
// Stage 3: LayerNorm(out) * gate, then @ Wo -> f32 output [p][128].
// =====================================================================
__global__ void __launch_bounds__(256)
trimul_stage3(const u16* __restrict__ outb, const u16* __restrict__ gateO,
              const float* __restrict__ on_g, const float* __restrict__ on_b,
              const float* __restrict__ Wo, float* __restrict__ out) {
    __shared__ u16 hs[128][136];
    __shared__ u16 wt[128][136];

    const int tid   = threadIdx.x;
    const int lane  = tid & 31;
    const int wv    = tid >> 5;
    const int wbase = wv * 16;
    const size_t p_base = (size_t)blockIdx.x * 128;

    // ---- LayerNorm over bf16 + gate multiply ----
    {
        const int row  = tid >> 1;
        const int half = tid & 1;
        const u16* orow = outb  + (p_base + row) * DIM + half * 64;
        const u16* grow = gateO + (p_base + row) * DIM + half * 64;
        uint4 raw[8], graw[8];
        float s = 0.f, sq = 0.f;
#pragma unroll
        for (int i = 0; i < 8; ++i) {
            raw[i]  = ((const uint4*)orow)[i];
            graw[i] = ((const uint4*)grow)[i];
            u32 rr[4] = {raw[i].x, raw[i].y, raw[i].z, raw[i].w};
#pragma unroll
            for (int q = 0; q < 4; ++q) {
                float v0 = bf2f((u16)(rr[q] & 0xFFFFu));
                float v1 = bf2f((u16)(rr[q] >> 16));
                s  += v0 + v1;
                sq += v0 * v0 + v1 * v1;
            }
        }
        s  += __shfl_xor(s, 1, 32);
        sq += __shfl_xor(sq, 1, 32);
        const float mu  = s * (1.f / 128.f);
        const float var = sq * (1.f / 128.f) - mu * mu;
        const float rs  = rsqrtf(var + LNEPS);
#pragma unroll
        for (int i = 0; i < 8; ++i) {
            u32 rr[4] = {raw[i].x,  raw[i].y,  raw[i].z,  raw[i].w};
            u32 gg[4] = {graw[i].x, graw[i].y, graw[i].z, graw[i].w};
#pragma unroll
            for (int q = 0; q < 4; ++q) {
                const int c = half * 64 + i * 8 + q * 2;
                float v0 = bf2f((u16)(rr[q] & 0xFFFFu));
                float v1 = bf2f((u16)(rr[q] >> 16));
                float g0 = bf2f((u16)(gg[q] & 0xFFFFu));
                float g1 = bf2f((u16)(gg[q] >> 16));
                u32 pk = pack2bf(((v0 - mu) * rs * on_g[c]     + on_b[c])     * g0,
                                 ((v1 - mu) * rs * on_g[c + 1] + on_b[c + 1]) * g1);
                *(u32*)&hs[row][c] = pk;     // one b32 store per pair
            }
        }
    }
    __syncthreads();
    load_w(Wo, wt, tid);
    __syncthreads();

    const int mh = (lane >> 4) * 8;
#pragma unroll 1
    for (int nt = 0; nt < 8; ++nt) {
        f32x8 c = mm_tile16(hs, wbase, wt, nt * 16, lane);
#pragma unroll
        for (int vv = 0; vv < 8; ++vv) {
            const size_t prow = p_base + wbase + mh + vv;
            out[prow * DIM + nt * 16 + (lane & 15)] = c[vv];
        }
    }
}

// =====================================================================
extern "C" void kernel_launch(void* const* d_in, const int* in_sizes, int n_in,
                              void* d_out, int out_size, void* d_ws, size_t ws_size,
                              hipStream_t stream) {
    const float* x    = (const float*)d_in[0];
    const float* mask = (const float*)d_in[1];
    const float* ln_g = (const float*)d_in[2];
    const float* ln_b = (const float*)d_in[3];
    const float* Wl   = (const float*)d_in[4];
    const float* Wr   = (const float*)d_in[5];
    const float* Wlg  = (const float*)d_in[6];
    const float* Wrg  = (const float*)d_in[7];
    const float* Wog  = (const float*)d_in[8];
    const float* on_g = (const float*)d_in[9];
    const float* on_b = (const float*)d_in[10];
    const float* Wo   = (const float*)d_in[11];

    // workspace: 4 x 64MB bf16 buffers
    u16* leftT  = (u16*)d_ws;                       // [128][SS]
    u16* rightT = leftT  + (size_t)DIM * SS;        // [128][SS]
    u16* gateO  = rightT + (size_t)DIM * SS;        // [SS][128]
    u16* outb   = gateO  + (size_t)SS * DIM;        // [SS][128]

    trimul_stage1<<<SS / 128, 256, 0, stream>>>(
        x, mask, ln_g, ln_b, Wl, Wr, Wlg, Wrg, Wog, leftT, rightT, gateO);
    trimul_stage2<<<DIM * 32, 128, 0, stream>>>(leftT, rightT, outb);
    trimul_stage3<<<SS / 128, 256, 0, stream>>>(
        outb, gateO, on_g, on_b, Wo, (float*)d_out);
}
